// MultiHeadAttention_26731876450511
// MI455X (gfx1250) — compile-verified, matrix operand order checked by reading, not run on hardware
//
#include <hip/hip_runtime.h>

#define D_MODEL 1024
#define NUM_HEADS 16
#define D_K 64
#define BATCH 2
#define SEQ 2048

typedef __attribute__((ext_vector_type(16))) __bf16 v16bf;
typedef __attribute__((ext_vector_type(8)))  float  v8f;
typedef __attribute__((ext_vector_type(4)))  unsigned int v4u;
typedef __attribute__((ext_vector_type(8)))  unsigned int v8u;

// ---------------------------------------------------------------------------
// WMMA wrapper: D = A(16x32 bf16) * B(32x16 bf16) + C(16x16 f32)
// ---------------------------------------------------------------------------
static __device__ inline v8f wmma_bf16(v16bf a, v16bf b, v8f c) {
  return __builtin_amdgcn_wmma_f32_16x16x32_bf16(
      /*neg_a=*/false, a, /*neg_b=*/false, b,
      /*c_mod=*/(short)0, c, /*reuse_a=*/false, /*reuse_b=*/false);
}

// ---------------------------------------------------------------------------
// CDNA5 async global->LDS copy (ASYNCcnt-tracked; ISA 10 / 15.18.3).
// lds_off is a byte offset into the dynamic LDS segment (base 0: no static
// __shared__ in the kernels that use this).
// ---------------------------------------------------------------------------
static __device__ inline void async_copy_b128(unsigned lds_off, const void* gaddr) {
  asm volatile("global_load_async_to_lds_b128 %0, %1, off"
               :: "v"(lds_off), "v"(gaddr) : "memory");
}
static __device__ inline void wait_async_le4() {
  asm volatile("s_wait_asynccnt 0x4" ::: "memory");
}
static __device__ inline void wait_async_le0() {
  asm volatile("s_wait_asynccnt 0x0" ::: "memory");
}

// ---------------------------------------------------------------------------
// CDNA5 Tensor Data Mover: 2-D tile DMA global->LDS (TENSORcnt; ISA ch.8).
// D# group0 (4 SGPRs): count=1 | lds_addr | global_addr[56:0] | type=2.
// D# group1 (8 SGPRs): data_size=2B, tensor_dim0/1, tile_dim0/1, dim0 stride.
// All inputs wave-uniform -> "s" constraints are legal.
// ---------------------------------------------------------------------------
static __device__ inline void tdm_load_2d(unsigned lds_off, const void* gaddr,
                                          unsigned tile_d0, unsigned tile_d1,
                                          unsigned tensor_d0, unsigned tensor_d1,
                                          unsigned long long stride0_elems) {
  unsigned long long ga = (unsigned long long)(size_t)gaddr;
  v4u g0;
  g0[0] = 1u;                                               // count=1, user D#
  g0[1] = lds_off;                                          // lds_addr (bytes)
  g0[2] = (unsigned)(ga & 0xFFFFFFFFu);                     // global_addr[31:0]
  g0[3] = (unsigned)((ga >> 32) & 0x01FFFFFFu) | (2u << 30);// addr[56:32]|type=2
  v8u g1;
  g1[0] = (1u << 16);                                       // data_size=1 (2B)
  g1[1] = (tensor_d0 & 0xFFFFu) << 16;                      // tensor_dim0 lo16
  g1[2] = (tensor_d0 >> 16) | ((tensor_d1 & 0xFFFFu) << 16);
  g1[3] = (tensor_d1 >> 16) | (tile_d0 << 16);              // tile_dim0
  g1[4] = tile_d1;                                          // tile_dim1, dim2=0
  g1[5] = (unsigned)(stride0_elems & 0xFFFFFFFFu);          // dim0 stride lo
  g1[6] = (unsigned)((stride0_elems >> 32) & 0xFFFFu);      // stride hi
  g1[7] = 0u;                                               // dim1 stride unused
  asm volatile("tensor_load_to_lds %0, %1" :: "s"(g0), "s"(g1) : "memory");
}
static __device__ inline void wait_tensor_le1() {
  asm volatile("s_wait_tensorcnt 0x1" ::: "memory");
}
static __device__ inline void wait_tensor_le0() {
  asm volatile("s_wait_tensorcnt 0x0" ::: "memory");
}

// ---------------------------------------------------------------------------
// Fragment loaders. 16-bit A-matrix 16x32 layout (ISA 7.12.2):
//   lane<16 : row M=lane,      elems 0..7 = K k0+0..7,  elems 8..15 = K k0+16..23
//   lane>=16: row M=lane-16,   elems 0..7 = K k0+8..15, elems 8..15 = K k0+24..31
// B-fragment (KxN) uses the same per-lane pattern with lane = N and the
// source row being row N of a row-major [N][K] matrix.
// ---------------------------------------------------------------------------
static __device__ inline v16bf frag_from_f32(const float* __restrict__ base,
                                             int ld, int row0, int k0, int lane) {
  int r  = lane & 15;
  int kk = k0 + ((lane & 16) ? 8 : 0);
  const float* p = base + (size_t)(row0 + r) * ld + kk;
  v16bf f;
#pragma unroll
  for (int i = 0; i < 8; ++i) f[i] = (__bf16)p[i];
#pragma unroll
  for (int i = 0; i < 8; ++i) f[i + 8] = (__bf16)p[i + 16];
  return f;
}

static __device__ inline v16bf frag_from_bf16(const __bf16* __restrict__ base,
                                              int ld, int row0, int k0, int lane) {
  int r  = lane & 15;
  int kk = k0 + ((lane & 16) ? 8 : 0);
  const __bf16* p = base + (size_t)(row0 + r) * ld + kk;
  v16bf f;
#pragma unroll
  for (int i = 0; i < 8; ++i) f[i] = p[i];
#pragma unroll
  for (int i = 0; i < 8; ++i) f[i + 8] = p[i + 16];
  return f;
}

// Store one 16x16 f32 C/D tile (+bias) to the selected output layout.
static __device__ inline void store_tile(v8f acc, int m0, int n0, int lane,
                                         const float* __restrict__ bias,
                                         __bf16* __restrict__ outb,
                                         float* __restrict__ outf, int mode) {
  int n  = n0 + (lane & 15);
  int ib = (lane & 16) ? 8 : 0;
  float bv = bias[n];
  int h = n >> 6, dk = n & 63;
#pragma unroll
  for (int r = 0; r < 8; ++r) {
    int m = m0 + ib + r;
    int bb = m / SEQ, s = m % SEQ;
    float v = acc[r] + bv;
    if (mode == 0)
      outb[(((size_t)bb * NUM_HEADS + h) * SEQ + s) * D_K + dk] = (__bf16)v;
    else if (mode == 1)
      outb[(((size_t)bb * NUM_HEADS + h) * D_K + dk) * SEQ + s] = (__bf16)v;
    else
      outf[(size_t)m * D_MODEL + n] = v;
  }
}

// ---------------------------------------------------------------------------
// Projection GEMM: Y = X @ W^T + bias.  M = B*S, N = K = 1024.
// Block = 128 thr = 4 waves covering 64(M) x 64(N); each wave owns a 32x32
// register tile: 2 A-frags x 2 B-frags -> 4 WMMAs per 32-deep k-step.
// ---------------------------------------------------------------------------
__global__ void proj_kernel(const float* __restrict__ X, const float* __restrict__ W,
                            const float* __restrict__ bias,
                            __bf16* __restrict__ outb, float* __restrict__ outf,
                            int mode) {
  int wave = threadIdx.x >> 5;
  int lane = threadIdx.x & 31;
  int m0 = blockIdx.x * 64 + (wave & 1) * 32;
  int n0 = blockIdx.y * 64 + (wave >> 1) * 32;

  v8f acc00 = {}, acc01 = {}, acc10 = {}, acc11 = {};
  for (int k0 = 0; k0 < D_MODEL; k0 += 32) {
    if (k0 + 64 < D_MODEL) {           // hint next k-block into near cache
      __builtin_prefetch(X + (size_t)(m0 + (lane & 15)) * D_MODEL + k0 + 64, 0, 3);
      __builtin_prefetch(W + (size_t)(n0 + (lane & 15)) * D_MODEL + k0 + 64, 0, 3);
    }
    v16bf a0 = frag_from_f32(X, D_MODEL, m0,      k0, lane);
    v16bf a1 = frag_from_f32(X, D_MODEL, m0 + 16, k0, lane);
    v16bf b0 = frag_from_f32(W, D_MODEL, n0,      k0, lane);
    v16bf b1 = frag_from_f32(W, D_MODEL, n0 + 16, k0, lane);
    acc00 = wmma_bf16(a0, b0, acc00);
    acc01 = wmma_bf16(a0, b1, acc01);
    acc10 = wmma_bf16(a1, b0, acc10);
    acc11 = wmma_bf16(a1, b1, acc11);
  }
  store_tile(acc00, m0,      n0,      lane, bias, outb, outf, mode);
  store_tile(acc01, m0,      n0 + 16, lane, bias, outb, outf, mode);
  store_tile(acc10, m0 + 16, n0,      lane, bias, outb, outf, mode);
  store_tile(acc11, m0 + 16, n0 + 16, lane, bias, outb, outf, mode);
}

// ---------------------------------------------------------------------------
// Attention LDS budget (dynamic segment, byte offsets from 0):
//   sP   : 16 x SEQ f32 scores                         [0, 131072)
//   red  : 16 x 8 f32 reduction scratch                [131072, 131584)
//   kstg : 4 waves x 2 x 2KB K-tile async staging      [131584, 147968)
//   vstg : 4 waves x 2 x 1KB V-tile TDM staging        [147968, 156160)
// ---------------------------------------------------------------------------
#define SP_FLOATS   (16 * SEQ)
#define RED_FLOATS  (16 * 8)
#define KSTG_BASE   ((SP_FLOATS + RED_FLOATS) * 4)
#define KSTG_WAVE   4096
#define VSTG_BASE   (KSTG_BASE + 4 * KSTG_WAVE)
#define VSTG_WAVE   2048
#define SMEM_BYTES  (VSTG_BASE + 4 * VSTG_WAVE)

// Stage one 16x64 bf16 K-tile (2 KB) into LDS: 4 async b128 issues per wave.
static __device__ inline void stage_ktile(const __bf16* __restrict__ Kh, int j0,
                                          unsigned lds_off, int lane) {
#pragma unroll
  for (int c = 0; c < 4; ++c) {
    int idx = c * 32 + lane;            // 128 chunks of 16 B
    int row = idx >> 3, seg = idx & 7;
    async_copy_b128(lds_off + (unsigned)idx * 16,
                    Kh + (size_t)(j0 + row) * D_K + seg * 8);
  }
}

// ---------------------------------------------------------------------------
// Attention: one block (4 waves) per (head, 16-query-row block).
//   phase 1: S = scale*Q K^T via WMMA; K tiles double-buffered into LDS with
//            global_load_async_to_lds_b128 + s_wait_asynccnt pipelining.
//   phase 2: LDS-resident softmax (float4 sweeps), write attn (b128 stores).
//   phase 3: O = P V via WMMA; V^T tiles double-buffered into LDS with the
//            Tensor Data Mover (tensor_load_to_lds + s_wait_tensorcnt).
// ---------------------------------------------------------------------------
__global__ void attn_kernel(const __bf16* __restrict__ Qb, const __bf16* __restrict__ Kb,
                            const __bf16* __restrict__ Vt,
                            const unsigned char* __restrict__ mask,
                            float* __restrict__ attn, float* __restrict__ heads) {
  extern __shared__ float smem[];
  float* sP  = smem;                  // [16][SEQ]
  float* red = smem + SP_FLOATS;      // [16][8]

  int wave = threadIdx.x >> 5, lane = threadIdx.x & 31;
  int bh = blockIdx.y;
  int b = bh >> 4, h = bh & 15;
  int row0 = blockIdx.x * 16;

  const __bf16* Qh = Qb + (size_t)bh * SEQ * D_K;
  const __bf16* Kh = Kb + (size_t)bh * SEQ * D_K;
  const __bf16* Vh = Vt + (size_t)bh * D_K * SEQ;   // [dk][s]
  const unsigned char* mb = mask + (size_t)b * SEQ * SEQ;

  // ---- phase 1: scores -> LDS -------------------------------------------
  v16bf qa0 = frag_from_bf16(Qh, D_K, row0, 0,  lane);
  v16bf qa1 = frag_from_bf16(Qh, D_K, row0, 32, lane);

  unsigned kst0 = KSTG_BASE + (unsigned)wave * KSTG_WAVE;   // two 2 KB buffers
  const __bf16* sK0 = (const __bf16*)((const char*)smem + kst0);
  const __bf16* sK1 = (const __bf16*)((const char*)smem + kst0 + 2048);

  int jj = lane & 15;
  int ib = (lane & 16) ? 8 : 0;
  int jbase = wave * (SEQ / 4);                       // 512 contiguous cols/wave

  stage_ktile(Kh, jbase, kst0, lane);
  for (int t = 0; t < SEQ / (16 * 4); ++t) {          // 32 col-tiles per wave
    int j0 = jbase + t * 16;
    if (t + 1 < SEQ / (16 * 4)) {
      stage_ktile(Kh, j0 + 16, ((t + 1) & 1) ? kst0 + 2048 : kst0, lane);
      wait_async_le4();                               // tile t resident
    } else {
      wait_async_le0();
    }
    const __bf16* sk = (t & 1) ? sK1 : sK0;
    v16bf kb0 = frag_from_bf16(sk, D_K, 0, 0,  lane);
    v16bf kb1 = frag_from_bf16(sk, D_K, 0, 32, lane);
    v8f acc = {};
    acc = wmma_bf16(qa0, kb0, acc);
    acc = wmma_bf16(qa1, kb1, acc);
#pragma unroll
    for (int r = 0; r < 8; ++r) {
      int ii = ib + r;
      float v = acc[r] * 0.125f;                      // 1/sqrt(64)
      if (!mb[(size_t)(row0 + ii) * SEQ + (j0 + jj)]) v = -10000.0f;
      sP[ii * SEQ + j0 + jj] = v;
    }
  }
  __syncthreads();

  // ---- phase 2: softmax in LDS (float4 sweeps), write attn --------------
  int row = threadIdx.x >> 3, seg = threadIdx.x & 7;  // 8 threads per row
  float* rp = sP + row * SEQ + seg * 256;
  float4* rp4 = (float4*)rp;
  float mx = -3.0e38f;
  for (int c = 0; c < 64; ++c) {
    float4 u = rp4[c];
    mx = fmaxf(mx, fmaxf(fmaxf(u.x, u.y), fmaxf(u.z, u.w)));
  }
  red[row * 8 + seg] = mx;
  __syncthreads();
  float rm = red[row * 8];
  for (int i = 1; i < 8; ++i) rm = fmaxf(rm, red[row * 8 + i]);
  __syncthreads();
  float sum = 0.f;
  for (int c = 0; c < 64; ++c) {
    float4 u = rp4[c];
    u.x = __expf(u.x - rm); u.y = __expf(u.y - rm);
    u.z = __expf(u.z - rm); u.w = __expf(u.w - rm);
    rp4[c] = u;
    sum += (u.x + u.y) + (u.z + u.w);
  }
  red[row * 8 + seg] = sum;
  __syncthreads();
  float rs = 0.f;
  for (int i = 0; i < 8; ++i) rs += red[row * 8 + i];
  float inv = 1.0f / rs;
  float4* arow4 = (float4*)(attn + ((size_t)bh * SEQ + row0 + row) * SEQ + seg * 256);
  for (int c = 0; c < 64; ++c) {
    float4 u = rp4[c];
    u.x *= inv; u.y *= inv; u.z *= inv; u.w *= inv;
    rp4[c] = u;
    arow4[c] = u;
  }
  __syncthreads();

  // ---- phase 3: O = P V, V^T tiles via TDM double buffering -------------
  v8f oacc = {};
  int dk0 = wave * 16;
  unsigned vst0 = VSTG_BASE + (unsigned)wave * VSTG_WAVE;   // two 1 KB buffers
  const __bf16* sV0 = (const __bf16*)((const char*)smem + vst0);
  const __bf16* sV1 = (const __bf16*)((const char*)smem + vst0 + 1024);
  const __bf16* VhW = Vh + (size_t)dk0 * SEQ;               // rows dk0..dk0+15

  tdm_load_2d(vst0, VhW, /*tile*/32, 16, /*tensor*/SEQ, 16, SEQ);
  for (int t = 0; t < SEQ / 32; ++t) {
    int j0 = t * 32;
    if (j0 + 32 < SEQ) {
      tdm_load_2d(((t + 1) & 1) ? vst0 + 1024 : vst0, VhW + (j0 + 32),
                  32, 16, (unsigned)(SEQ - (j0 + 32)), 16, SEQ);
      wait_tensor_le1();                              // tile t resident
    } else {
      wait_tensor_le0();
    }
    const __bf16* sv = (t & 1) ? sV1 : sV0;           // staged [16][32] bf16
    v16bf pa = frag_from_f32(sP, SEQ, 0, j0, lane);   // P from LDS (cvt bf16)
    v16bf vb = frag_from_bf16(sv, 32, 0, 0, lane);
    oacc = wmma_bf16(pa, vb, oacc);
  }
  int n = lane & 15;
#pragma unroll
  for (int r = 0; r < 8; ++r) {
    int i = ib + r;
    heads[((size_t)b * SEQ + row0 + i) * D_MODEL + h * D_K + dk0 + n] = oacc[r];
  }
}

// ---------------------------------------------------------------------------
extern "C" void kernel_launch(void* const* d_in, const int* in_sizes, int n_in,
                              void* d_out, int out_size, void* d_ws, size_t ws_size,
                              hipStream_t stream) {
  (void)in_sizes; (void)n_in; (void)out_size; (void)ws_size;
  const float* q  = (const float*)d_in[0];
  const float* k  = (const float*)d_in[1];
  const float* v  = (const float*)d_in[2];
  const unsigned char* mask = (const unsigned char*)d_in[3];
  const float* Wq = (const float*)d_in[4];
  const float* bq = (const float*)d_in[5];
  const float* Wk = (const float*)d_in[6];
  const float* bk = (const float*)d_in[7];
  const float* Wv = (const float*)d_in[8];
  const float* bv = (const float*)d_in[9];
  const float* Wo = (const float*)d_in[10];
  const float* bo = (const float*)d_in[11];

  float* out  = (float*)d_out;                               // (B,S,D)
  float* attn = out + (size_t)BATCH * SEQ * D_MODEL;         // (B,H,S,S)

  size_t nqkv = (size_t)BATCH * NUM_HEADS * SEQ * D_K;       // 4 Mi elems
  __bf16* Qb = (__bf16*)d_ws;
  __bf16* Kb = Qb + nqkv;
  __bf16* Vt = Kb + nqkv;
  float* heads = (float*)(Vt + nqkv);                        // (B,S,D) f32

  dim3 pgrid(BATCH * SEQ / 64, D_MODEL / 64);                // 64 x 16
  proj_kernel<<<pgrid, 128, 0, stream>>>(q, Wq, bq, Qb, nullptr, 0);
  proj_kernel<<<pgrid, 128, 0, stream>>>(k, Wk, bk, Kb, nullptr, 0);
  proj_kernel<<<pgrid, 128, 0, stream>>>(v, Wv, bv, Vt, nullptr, 1);

  attn_kernel<<<dim3(SEQ / 16, BATCH * NUM_HEADS), 128, SMEM_BYTES, stream>>>(
      Qb, Kb, Vt, mask, attn, heads);

  proj_kernel<<<pgrid, 128, 0, stream>>>(heads, Wo, bo, nullptr, out, 2);
}
